// CausalSelfAttention_87179246174510
// MI455X (gfx1250) — matrix operand order repaired and checked by reading, not compiled, not run on hardware
//
#include <hip/hip_runtime.h>

typedef __bf16 v16bf __attribute__((ext_vector_type(16)));
typedef __bf16 bf16x8 __attribute__((ext_vector_type(8)));
typedef float  v8f   __attribute__((ext_vector_type(8)));

#define WMMA_BF16(a, b, c) \
  __builtin_amdgcn_wmma_f32_16x16x32_bf16(false, (a), false, (b), (short)0, (c), false, false)

// --- TDM support (6-arg clang-23 form: g0 u32x4, g1 i32x8, i32x4, i32x4,
// i32x8, cpol) — arity confirmed by the round-2 diagnostic. -----------------
#if defined(__HIP_DEVICE_COMPILE__) && defined(__has_builtin)
#if __has_builtin(__builtin_amdgcn_tensor_load_to_lds) && \
    __has_builtin(__builtin_amdgcn_s_wait_tensorcnt)
#define HAVE_TDM 1
#endif
#endif

#ifdef HAVE_TDM
typedef unsigned int u32x4 __attribute__((ext_vector_type(4)));
typedef int          i32x4 __attribute__((ext_vector_type(4)));
typedef int          i32x8 __attribute__((ext_vector_type(8)));

// 2D bf16 tile load Global -> LDS via Tensor Data Mover.
// D# bitfields per CDNA5 ISA 8.3/8.4: group0 = {count=1, lds_addr,
// global_addr[56:0], type=2}; group1 = {wg_mask=0, data_size=1 (2B),
// tensor_dim0/1 = tile_w/tile_h (tiles never OOB here), tile_dim0/1,
// tensor_dim0_stride = row_stride}.  Remaining groups zero (2D tile:
// tile_dim2/3/4 = 0 means unused).
__device__ __forceinline__ void tdm_load_2d_bf16(const __bf16* gsrc,
                                                 unsigned lds_byte_addr,
                                                 unsigned tile_w, unsigned tile_h,
                                                 unsigned row_stride_elems) {
  unsigned long long ga = (unsigned long long)(const void*)gsrc;
  u32x4 g0;
  g0[0] = 1u;                                              // count=1 (valid user D#)
  g0[1] = lds_byte_addr;                                   // LDS byte address
  g0[2] = (unsigned)(ga & 0xFFFFFFFFu);                    // global_addr[31:0]
  g0[3] = (unsigned)((ga >> 32) & 0x01FFFFFFu) | (2u << 30); // [56:32] + type=2
  i32x8 g1;
  g1[0] = (int)(1u << 16);                                 // data_size=1 (2 bytes)
  g1[1] = (int)((tile_w & 0xFFFFu) << 16);                 // tensor_dim0[15:0]
  g1[2] = (int)(((tile_w >> 16) & 0xFFFFu) | ((tile_h & 0xFFFFu) << 16)); // dim0 hi | dim1 lo
  g1[3] = (int)(((tile_h >> 16) & 0xFFFFu) | ((tile_w & 0xFFFFu) << 16)); // dim1 hi | tile_dim0
  g1[4] = (int)(tile_h & 0xFFFFu);                         // tile_dim1 (tile_dim2 = 0)
  g1[5] = (int)row_stride_elems;                           // tensor_dim0_stride[31:0]
  g1[6] = 0;                                               // stride hi / dim1_stride lo
  g1[7] = 0;
  i32x4 gz4 = {0, 0, 0, 0};
  i32x8 gz8 = {0, 0, 0, 0, 0, 0, 0, 0};
  __builtin_amdgcn_tensor_load_to_lds(g0, g1, gz4, gz4, gz8, 0);
}
#endif

// ---------------------------------------------------------------------------
// Fragment loaders (wave32, 16x16x32 bf16 WMMA).
// A 16x32: lane=(half,m), element i -> k = (i&7) + (i>>3)*16 + half*8.
// B 32x16: lane=(half,n), element i -> k = (i&7) + (i>>3)*16 + half*8.
// `line` points at k=0 of this lane's row/column; K contiguous in memory.
// ---------------------------------------------------------------------------
__device__ __forceinline__ v16bf fragA_from_line(const __bf16* line, int half) {
  const bf16x8* p = (const bf16x8*)line;
  bf16x8 lo = p[half];
  bf16x8 hi = p[2 + half];
  v16bf r;
#pragma unroll
  for (int i = 0; i < 8; ++i) { r[i] = lo[i]; r[i + 8] = hi[i]; }
  return r;
}

__device__ __forceinline__ v16bf fragB_from_line(const __bf16* line, int half) {
  const bf16x8* p = (const bf16x8*)line;
  bf16x8 lo = p[half];
  bf16x8 hi = p[2 + half];
  v16bf r;
#pragma unroll
  for (int i = 0; i < 8; ++i) { r[i] = lo[i]; r[i + 8] = hi[i]; }
  return r;
}

// ---------------------------------------------------------------------------
// fp32 -> bf16 elementwise conversion
// ---------------------------------------------------------------------------
__global__ __launch_bounds__(256) void f32_to_bf16(const float* __restrict__ src,
                                                   __bf16* __restrict__ dst,
                                                   long n) {
  long i = (long)blockIdx.x * blockDim.x + threadIdx.x;
  if (i < n) dst[i] = (__bf16)src[i];
}

// ---------------------------------------------------------------------------
// Tiled bf16 GEMM with bias: C[M,N] = A[M,K] @ B[K,N] + bias[N]
// Block tile 128x128, K-step 32, 256 threads (8 waves), wave tile 32x64.
// A tile staged by the Tensor Data Mover (one D# per K-step); B tile staged
// manually (TDM cannot transpose, and fragB wants K-contiguous columns).
// ---------------------------------------------------------------------------
template <typename OutT>
__global__ __launch_bounds__(256) void gemm_bias_wmma(
    const __bf16* __restrict__ A, const __bf16* __restrict__ B,
    const float* __restrict__ bias, OutT* __restrict__ C,
    int N, int K) {
  __shared__ __bf16 As[128 * 32];   // [row][k]
  __shared__ __bf16 Bs[128 * 32];   // [n][k] (transposed on load)

  const int tid  = threadIdx.x;
  const int lane = tid & 31;
  const int wave = tid >> 5;
  const int n16  = lane & 15;
  const int half = lane >> 4;
  const int wm   = wave & 3;   // 32-row band
  const int wn   = wave >> 2;  // 64-col band
  const long rowStart = (long)blockIdx.y * 128;
  const long colStart = (long)blockIdx.x * 128;

  v8f zero = {0.f, 0.f, 0.f, 0.f, 0.f, 0.f, 0.f, 0.f};
  v8f acc[2][4];
#pragma unroll
  for (int i = 0; i < 2; ++i)
#pragma unroll
    for (int j = 0; j < 4; ++j) acc[i][j] = zero;

  const int bk = tid >> 3;          // 0..31   (B loader k)
  const int bn = (tid & 7) * 16;    // 0..112  (B loader n chunk)
#ifndef HAVE_TDM
  const int ar = tid >> 1;          // 0..127  (A loader row, fallback path)
  const int ak = (tid & 1) * 16;    // 0/16
#endif

  for (int k0 = 0; k0 < K; k0 += 32) {
    // ---- B tile: 32x128 global -> LDS transposed [n][k] -------------------
    const bf16x8* bsrc = (const bf16x8*)(B + (long)(k0 + bk) * N + colStart + bn);
    bf16x8 b0 = bsrc[0], b1 = bsrc[1];
#pragma unroll
    for (int e = 0; e < 8; ++e) {
      Bs[(bn + e) * 32 + bk]     = b0[e];
      Bs[(bn + 8 + e) * 32 + bk] = b1[e];
    }
    if (k0 + 32 < K)  // prefetch next B tile -> global_prefetch_b8
      __builtin_prefetch(B + (long)(k0 + 32 + bk) * N + colStart + bn, 0, 1);

    // ---- A tile: 128x32 global -> LDS [row][k] ----------------------------
#ifdef HAVE_TDM
    if (wave == 0) {
      tdm_load_2d_bf16(A + rowStart * K + k0,
                       (unsigned)(size_t)(const void*)As,
                       /*tile_w=*/32, /*tile_h=*/128, /*row_stride=*/(unsigned)K);
      __builtin_amdgcn_s_wait_tensorcnt(0);
    }
#else
    {
      const bf16x8* asrc = (const bf16x8*)(A + (rowStart + ar) * K + k0 + ak);
      bf16x8 a0 = asrc[0], a1 = asrc[1];
      *(bf16x8*)(As + ar * 32 + ak)     = a0;
      *(bf16x8*)(As + ar * 32 + ak + 8) = a1;
    }
#endif
    __syncthreads();

    v16bf af[2], bfr[4];
#pragma unroll
    for (int i = 0; i < 2; ++i)
      af[i] = fragA_from_line(As + (wm * 32 + i * 16 + n16) * 32, half);
#pragma unroll
    for (int j = 0; j < 4; ++j)
      bfr[j] = fragB_from_line(Bs + (wn * 64 + j * 16 + n16) * 32, half);
#pragma unroll
    for (int i = 0; i < 2; ++i)
#pragma unroll
      for (int j = 0; j < 4; ++j)
        acc[i][j] = WMMA_BF16(af[i], bfr[j], acc[i][j]);
    __syncthreads();
  }

  // Epilogue: C-layout row = r + 8*half, col = n16 within each 16x16 tile
#pragma unroll
  for (int i = 0; i < 2; ++i) {
#pragma unroll
    for (int j = 0; j < 4; ++j) {
      const long col = colStart + wn * 64 + j * 16 + n16;
      const float bv = bias[col];
#pragma unroll
      for (int r = 0; r < 8; ++r) {
        const long row = rowStart + wm * 32 + i * 16 + r + 8 * half;
        C[row * N + col] = (OutT)(acc[i][j][r] + bv);
      }
    }
  }
}

// ---------------------------------------------------------------------------
// Flash attention (causal, no 1/sqrt(D) scaling — faithful to reference).
// qkv: [B*T, 2304] bf16 (q | k | v, each [H=12, D=64]).  attn: [B*T, 768] bf16.
// One block per (b, h, 128-row tile); 8 waves, each wave owns 16 query rows.
// ---------------------------------------------------------------------------
__global__ __launch_bounds__(256) void flash_attn_wmma(
    const __bf16* __restrict__ qkv, __bf16* __restrict__ attn, int T) {
  constexpr int C3 = 2304, C = 768, D = 64;
  __shared__ __bf16 Klds[32 * 64];      // [kv][d]
  __shared__ __bf16 Vt[64 * 32];        // [d][kv]  (transposed)
  __shared__ __bf16 Pb[8][16 * 32];     // per-wave P scratch [row][kv]

  const int tid  = threadIdx.x;
  const int lane = tid & 31;
  const int wave = tid >> 5;
  const int n16  = lane & 15;
  const int half = lane >> 4;

  const int b  = blockIdx.z;
  const int h  = blockIdx.y;
  const int rt = blockIdx.x;                 // 128-row tile index
  const long base = (long)b * T * C3;
  const int q0 = rt * 128 + wave * 16;       // wave's first query row

  // Q fragments (16x64) held in registers for the whole KV sweep
  const __bf16* qline = qkv + base + (long)(q0 + n16) * C3 + h * D;
  const v16bf Qf0 = fragA_from_line(qline, half);        // d = 0..31
  const v16bf Qf1 = fragA_from_line(qline + 32, half);   // d = 32..63

  v8f zero = {0.f, 0.f, 0.f, 0.f, 0.f, 0.f, 0.f, 0.f};
  v8f O[4];
#pragma unroll
  for (int j = 0; j < 4; ++j) O[j] = zero;
  float mrow[8], lrow[8];
#pragma unroll
  for (int r = 0; r < 8; ++r) { mrow[r] = -3e38f; lrow[r] = 0.f; }

  const int kv_end = rt * 128 + 128;
  for (int kv0 = 0; kv0 < kv_end; kv0 += 32) {
    // Cooperative staging: K block row-major, V block transposed
    {
      const int kr = tid >> 3;           // 0..31
      const int d0 = (tid & 7) * 8;      // 0..56
      bf16x8 k8 = *(const bf16x8*)(qkv + base + (long)(kv0 + kr) * C3 + C + h * D + d0);
      *(bf16x8*)(Klds + kr * 64 + d0) = k8;
      bf16x8 v8 = *(const bf16x8*)(qkv + base + (long)(kv0 + kr) * C3 + 2 * C + h * D + d0);
#pragma unroll
      for (int e = 0; e < 8; ++e) Vt[(d0 + e) * 32 + kr] = v8[e];
      if (kv0 + 32 < kv_end) {           // prefetch next KV block
        __builtin_prefetch(qkv + base + (long)(kv0 + 32 + kr) * C3 + C + h * D + d0, 0, 1);
        __builtin_prefetch(qkv + base + (long)(kv0 + 32 + kr) * C3 + 2 * C + h * D + d0, 0, 1);
      }
    }
    __syncthreads();

    if (kv0 <= q0 + 15) {   // uniform per wave -> EXEC stays all-ones at WMMA
      // S = Q @ K^T over 32 kv columns (two 16-wide column groups)
      v8f s0 = zero, s1 = zero;
      s0 = WMMA_BF16(Qf0, fragB_from_line(Klds + n16 * 64, half), s0);
      s0 = WMMA_BF16(Qf1, fragB_from_line(Klds + n16 * 64 + 32, half), s0);
      s1 = WMMA_BF16(Qf0, fragB_from_line(Klds + (16 + n16) * 64, half), s1);
      s1 = WMMA_BF16(Qf1, fragB_from_line(Klds + (16 + n16) * 64 + 32, half), s1);

      // Online softmax: rows live in 16-lane groups (C-layout), reduce via shfl_xor
      float scl[8];
      __bf16* pw = &Pb[wave][0];
#pragma unroll
      for (int r = 0; r < 8; ++r) {
        const int row = q0 + r + 8 * half;
        float v0 = (kv0 + n16 <= row)      ? s0[r] : -3e38f;   // causal mask
        float v1 = (kv0 + 16 + n16 <= row) ? s1[r] : -3e38f;
        float mx = fmaxf(v0, v1);
        mx = fmaxf(mx, __shfl_xor(mx, 1));
        mx = fmaxf(mx, __shfl_xor(mx, 2));
        mx = fmaxf(mx, __shfl_xor(mx, 4));
        mx = fmaxf(mx, __shfl_xor(mx, 8));
        const float newm = fmaxf(mrow[r], mx);
        const float p0 = __expf(v0 - newm);
        const float p1 = __expf(v1 - newm);
        float rs = p0 + p1;
        rs += __shfl_xor(rs, 1);
        rs += __shfl_xor(rs, 2);
        rs += __shfl_xor(rs, 4);
        rs += __shfl_xor(rs, 8);
        const float corr = __expf(mrow[r] - newm);
        lrow[r] = lrow[r] * corr + rs;
        mrow[r] = newm;
        scl[r]  = corr;
        pw[(r + 8 * half) * 32 + n16]      = (__bf16)p0;   // P -> LDS (reshape)
        pw[(r + 8 * half) * 32 + 16 + n16] = (__bf16)p1;
      }
#pragma unroll
      for (int j = 0; j < 4; ++j)
#pragma unroll
        for (int r = 0; r < 8; ++r) O[j][r] *= scl[r];

      // O += P(16x32) @ V(32x64); P re-read in A-fragment layout
      const v16bf Pf = fragA_from_line(&Pb[wave][n16 * 32], half);
#pragma unroll
      for (int j = 0; j < 4; ++j) {
        v16bf Vb = fragB_from_line(Vt + (j * 16 + n16) * 32, half);
        O[j] = WMMA_BF16(Pf, Vb, O[j]);
      }
    }
    __syncthreads();
  }

  // Normalize and store
  const long obase = (long)b * T * C;
#pragma unroll
  for (int r = 0; r < 8; ++r) {
    const float inv = 1.f / lrow[r];
    const long row = q0 + r + 8 * half;
#pragma unroll
    for (int j = 0; j < 4; ++j)
      attn[obase + row * C + h * D + j * 16 + n16] = (__bf16)(O[j][r] * inv);
  }
}

// ---------------------------------------------------------------------------
// Launch: convert -> QKV GEMM -> flash attention -> projection GEMM
// ---------------------------------------------------------------------------
extern "C" void kernel_launch(void* const* d_in, const int* in_sizes, int n_in,
                              void* d_out, int out_size, void* d_ws, size_t ws_size,
                              hipStream_t stream) {
  (void)in_sizes; (void)n_in; (void)out_size; (void)ws_size;
  const float* x      = (const float*)d_in[0];
  const float* w_attn = (const float*)d_in[1];
  const float* b_attn = (const float*)d_in[2];
  const float* w_proj = (const float*)d_in[3];
  const float* b_proj = (const float*)d_in[4];
  float* out = (float*)d_out;

  const int B = 8, T = 1024, C = 768, H = 12;
  const long BT = (long)B * T;

  char* ws = (char*)d_ws;
  __bf16* xb   = (__bf16*)ws;  ws += BT * C * 2;             // x in bf16
  __bf16* wab  = (__bf16*)ws;  ws += (long)C * 3 * C * 2;    // w_attn bf16
  __bf16* wpb  = (__bf16*)ws;  ws += (long)C * C * 2;        // w_proj bf16
  __bf16* qkv  = (__bf16*)ws;  ws += BT * 3 * C * 2;         // qkv bf16
  __bf16* attn = (__bf16*)ws;                                // pre-proj bf16

  long n = BT * C;
  f32_to_bf16<<<(unsigned)((n + 255) / 256), 256, 0, stream>>>(x, xb, n);
  n = (long)C * 3 * C;
  f32_to_bf16<<<(unsigned)((n + 255) / 256), 256, 0, stream>>>(w_attn, wab, n);
  n = (long)C * C;
  f32_to_bf16<<<(unsigned)((n + 255) / 256), 256, 0, stream>>>(w_proj, wpb, n);

  // qkv = xb @ w_attn + b_attn   (M=8192, N=2304, K=768)
  gemm_bias_wmma<__bf16><<<dim3(3 * C / 128, (unsigned)(BT / 128)), 256, 0, stream>>>(
      xb, wab, b_attn, qkv, 3 * C, C);

  // causal attention per (b, h, 128-row tile)
  flash_attn_wmma<<<dim3(T / 128, H, B), 256, 0, stream>>>(qkv, attn, T);

  // out = attn @ w_proj + b_proj   (M=8192, N=768, K=768), fp32 output
  gemm_bias_wmma<float><<<dim3(C / 128, (unsigned)(BT / 128)), 256, 0, stream>>>(
      attn, wpb, b_proj, out, C, C);
}
